// NeuralODE_1_8950711846032
// MI455X (gfx1250) — matrix operand order repaired and checked by reading, not compiled, not run on hardware
//
#include <hip/hip_runtime.h>

#define BLK   256
#define NH    100
#define HSTR  128          // padded K stride in halfs (4 k-tiles of 32)
#define NPAD  112          // padded N (7 n-tiles of 16)

typedef __attribute__((ext_vector_type(16))) _Float16 v16h;
typedef __attribute__((ext_vector_type(8)))  _Float16 v8h;
typedef __attribute__((ext_vector_type(8)))  float    v8f;

__device__ __forceinline__ float fast_tanh(float x) {
#if __has_builtin(__builtin_amdgcn_tanhf)
    return __builtin_amdgcn_tanhf(x);
#elif __has_builtin(__builtin_amdgcn_tanh_f32)
    return __builtin_amdgcn_tanh_f32(x);
#else
    // tanh(x) = 1 - 2/(exp2(2*log2e*x)+1); saturates correctly for |x| large
    float e = __builtin_amdgcn_exp2f(x * 2.8853900817779268f);
    return 1.0f - 2.0f * __builtin_amdgcn_rcpf(e + 1.0f);
#endif
}

__global__ __launch_bounds__(BLK) void neural_ode_kernel(
    const float* __restrict__ X,   const float* __restrict__ T,
    const float* __restrict__ pW1, const float* __restrict__ pb1,
    const float* __restrict__ pW2, const float* __restrict__ pb2,
    const float* __restrict__ pW3, const float* __restrict__ pb3,
    const float* __restrict__ oW1, const float* __restrict__ ob1,
    const float* __restrict__ oW2, const float* __restrict__ ob2,
    float* __restrict__ out, int Bn)
{
    // ---- LDS ----
    alignas(16) __shared__ _Float16 w2t[NPAD * HSTR];   // transposed pW2: [n][k], zero-padded
    alignas(16) __shared__ _Float16 Hs[BLK * HSTR];     // activations, one padded row per sample
    __shared__ float w1s[600];                          // pW1 [k][n]
    __shared__ float b1s[NPAD], b2s[NPAD];
    __shared__ float w3s[300], b3s[4];
    __shared__ float ow1s[300];                         // oW1 [k][n]
    __shared__ float ob1s[NH];
    __shared__ float ow2t[300];                         // [j][n]
    __shared__ float ob2s[4];
    __shared__ float Ts[16];

    const int tid = threadIdx.x;
    const long gid = (long)blockIdx.x * BLK + tid;

    // ---- Phase 0: stage weights ----
    for (int i = tid; i < NPAD * HSTR; i += BLK) {
        int n = i / HSTR, k = i % HSTR;
        float v = (n < NH && k < NH) ? pW2[k * NH + n] : 0.0f;
        w2t[i] = (_Float16)v;
    }
    for (int i = tid; i < 600; i += BLK) w1s[i] = pW1[i];
    for (int i = tid; i < NPAD; i += BLK) {
        b1s[i] = (i < NH) ? pb1[i] : 0.0f;
        b2s[i] = (i < NH) ? pb2[i] : 0.0f;
    }
    for (int i = tid; i < NH; i += BLK) ob1s[i] = ob1[i];
    for (int i = tid; i < 300; i += BLK) {
        w3s[i]  = pW3[i];
        ow1s[i] = oW1[i];
        int j = i / NH, n = i % NH;
        ow2t[i] = oW2[n * 3 + j];
    }
    if (tid < 4)  { b3s[tid] = (tid < 3) ? pb3[tid] : 0.0f;
                    ob2s[tid] = (tid < 3) ? ob2[tid] : 0.0f; }
    if (tid < 16) Ts[tid] = T[tid];
    __syncthreads();

    // ---- Phase 1: layer 1 (K=6, scalar per lane), f16 row into LDS ----
    {
        const float* xp = X + gid * 6;
        float x0 = xp[0], x1 = xp[1], x2 = xp[2], x3 = xp[3], x4 = xp[4], x5 = xp[5];
        _Float16* hrow = &Hs[tid * HSTR];
#pragma unroll 4
        for (int n = 0; n < NH; ++n) {
            float h = b1s[n] + x0 * w1s[n]       + x1 * w1s[100 + n]
                             + x2 * w1s[200 + n] + x3 * w1s[300 + n]
                             + x4 * w1s[400 + n] + x5 * w1s[500 + n];
            hrow[n] = (_Float16)fmaxf(h, 0.0f);
        }
        for (int n = NH; n < HSTR; ++n) hrow[n] = (_Float16)0.0f;
    }
    __syncthreads();

    // ---- Phase 2: layer 2 (100x100) via WMMA f16 -> f32 ----
    {
        const int lane = tid & 31;
        const int wv   = tid >> 5;
        const int ln   = lane & 15;      // column / row-in-tile index
        const int hh   = lane >> 4;      // lane half
        const v8f zero8 = {0.f, 0.f, 0.f, 0.f, 0.f, 0.f, 0.f, 0.f};

#pragma unroll 1
        for (int mt = 0; mt < 2; ++mt) {
            const int mrow = wv * 32 + mt * 16;
            v8f acc[7];
#pragma unroll
            for (int nt = 0; nt < 7; ++nt) acc[nt] = zero8;

#pragma unroll
            for (int kt = 0; kt < 4; ++kt) {
                // A fragment: lane half h holds K = kt*32 + h*8..+7 and kt*32+16+h*8..+7
                const _Float16* ap = &Hs[(mrow + ln) * HSTR + kt * 32 + hh * 8];
                v8h alo = *(const v8h*)ap;
                v8h ahi = *(const v8h*)(ap + 16);
                v16h a = __builtin_shufflevector(alo, ahi,
                          0,1,2,3,4,5,6,7,8,9,10,11,12,13,14,15);
#pragma unroll
                for (int nt = 0; nt < 7; ++nt) {
                    // B fragment: lane (col n) half h holds K = kt*32 + h*8..+7 and kt*32+16+h*8..+7
                    const _Float16* bp = &w2t[(nt * 16 + ln) * HSTR + kt * 32 + hh * 8];
                    v8h blo = *(const v8h*)bp;
                    v8h bhi = *(const v8h*)(bp + 16);
                    v16h b = __builtin_shufflevector(blo, bhi,
                              0,1,2,3,4,5,6,7,8,9,10,11,12,13,14,15);
                    acc[nt] = __builtin_amdgcn_wmma_f32_16x16x32_f16(
                        false, a, false, b, (short)0, acc[nt], false, false);
                }
            }
            // Epilogue: bias + ReLU, write f16 back over this wave's own rows
#pragma unroll
            for (int nt = 0; nt < 7; ++nt) {
                const int n = nt * 16 + ln;
                const float bias = b2s[n];
#pragma unroll
                for (int v = 0; v < 8; ++v) {
                    const int m = mrow + hh * 8 + v;
                    float val = fmaxf(acc[nt][v] + bias, 0.0f);
                    Hs[m * HSTR + n] = (n < NH) ? (_Float16)val : (_Float16)0.0f;
                }
            }
        }
    }
    __syncthreads();

    // ---- Phase 3: layer 3 (K=100, N=3) + RK4 ODE, fp32 per lane ----
    float ya = b3s[0], yb = b3s[1], yc = b3s[2];
    {
        const _Float16* h2 = &Hs[tid * HSTR];
#pragma unroll 4
        for (int k = 0; k < NH; ++k) {
            float hv = (float)h2[k];
            ya += hv * w3s[k * 3 + 0];
            yb += hv * w3s[k * 3 + 1];
            yc += hv * w3s[k * 3 + 2];
        }
    }

    auto odef = [&](float a, float b, float c, float& d0, float& d1, float& d2) {
        float s0 = 0.f, s1 = 0.f, s2 = 0.f;
#pragma unroll 4
        for (int n = 0; n < NH; ++n) {
            float t = fast_tanh(ob1s[n] + a * ow1s[n] + b * ow1s[100 + n] + c * ow1s[200 + n]);
            s0 += t * ow2t[n];
            s1 += t * ow2t[100 + n];
            s2 += t * ow2t[200 + n];
        }
        d0 = s0 + ob2s[0]; d1 = s1 + ob2s[1]; d2 = s2 + ob2s[2];
    };

    const long ostride = (long)Bn * 3;
    float* op0 = out + gid * 3;
    op0[0] = ya; op0[1] = yb; op0[2] = yc;

#pragma unroll 1
    for (int s = 0; s < 15; ++s) {
        const float dt = Ts[s + 1] - Ts[s];
        float k1a, k1b, k1c, k2a, k2b, k2c, k3a, k3b, k3c, k4a, k4b, k4c;
        odef(ya, yb, yc, k1a, k1b, k1c);
        odef(ya + 0.5f * dt * k1a, yb + 0.5f * dt * k1b, yc + 0.5f * dt * k1c, k2a, k2b, k2c);
        odef(ya + 0.5f * dt * k2a, yb + 0.5f * dt * k2b, yc + 0.5f * dt * k2c, k3a, k3b, k3c);
        odef(ya + dt * k3a, yb + dt * k3b, yc + dt * k3c, k4a, k4b, k4c);
        const float cdt = dt * (1.0f / 6.0f);
        ya += cdt * (k1a + 2.0f * k2a + 2.0f * k3a + k4a);
        yb += cdt * (k1b + 2.0f * k2b + 2.0f * k3b + k4b);
        yc += cdt * (k1c + 2.0f * k2c + 2.0f * k3c + k4c);
        float* op = out + (long)(s + 1) * ostride + gid * 3;
        op[0] = ya; op[1] = yb; op[2] = yc;
    }
}

extern "C" void kernel_launch(void* const* d_in, const int* in_sizes, int n_in,
                              void* d_out, int out_size, void* d_ws, size_t ws_size,
                              hipStream_t stream) {
    (void)n_in; (void)out_size; (void)d_ws; (void)ws_size;
    const float* X   = (const float*)d_in[0];
    const float* T   = (const float*)d_in[1];
    const float* pW1 = (const float*)d_in[2];
    const float* pb1 = (const float*)d_in[3];
    const float* pW2 = (const float*)d_in[4];
    const float* pb2 = (const float*)d_in[5];
    const float* pW3 = (const float*)d_in[6];
    const float* pb3 = (const float*)d_in[7];
    const float* oW1 = (const float*)d_in[8];
    const float* ob1 = (const float*)d_in[9];
    const float* oW2 = (const float*)d_in[10];
    const float* ob2 = (const float*)d_in[11];
    float* out = (float*)d_out;

    const int Bn = in_sizes[0] / 6;          // 524288
    dim3 grid(Bn / BLK), block(BLK);
    neural_ode_kernel<<<grid, block, 0, stream>>>(
        X, T, pW1, pb1, pW2, pb2, pW3, pb3, oW1, ob1, oW2, ob2, out, Bn);
}
